// SSMLayerZOH_10230612099620
// MI455X (gfx1250) — matrix operand order repaired and checked by reading, not compiled, not run on hardware
//
#include <hip/hip_runtime.h>

typedef __attribute__((ext_vector_type(16))) _Float16 v16h;
typedef __attribute__((ext_vector_type(8)))  float    v8f;
typedef int v4i_vec __attribute__((vector_size(16)));

#define B_SZ   16
#define CIN    256
#define NST    256
#define COUT   256
#define T_LEN  8192
#define TCH    64
#define NCHUNK (T_LEN / TCH)   // 128 chunks along T
#define XP     264             // pitch (elements) for LDS scan tiles (256 + 8 pad)
#define UP     40              // pitch (halves) for staged u slice (32 + 8 pad)
#define UBUF   (TCH * UP)      // 2560 halves per u stage buffer
#define CP     40              // pitch (halves) for staged CT slice
#define CBUF   (256 * CP)      // 10240 halves per CT stage buffer

#define AS1 __attribute__((address_space(1)))
#define AS3 __attribute__((address_space(3)))

#if __has_builtin(__builtin_amdgcn_global_load_async_to_lds_b128) && \
    __has_builtin(__builtin_amdgcn_s_wait_asynccnt)
#define HAVE_ASYNC_LDS 1
#else
#define HAVE_ASYNC_LDS 0
#endif

union Frag16 { v16h v; uint4 q[2]; };

// A-fragment (16x32 f16): lane half 'hi' holds K in {hi*8..+7} and {16+hi*8..+7}
__device__ __forceinline__ v16h ld_fragA(const _Float16* p0) {
  Frag16 f;
  f.q[0] = *(const uint4*)(p0);
  f.q[1] = *(const uint4*)(p0 + 16);
  return f.v;
}
// B-fragment (32x16 f16): lane half 'hi' holds K in {hi*8..+7} and {16+hi*8..+7} (caller adds hi*8)
__device__ __forceinline__ v16h ld_fragB(const _Float16* p0) {
  Frag16 f;
  f.q[0] = *(const uint4*)(p0);
  f.q[1] = *(const uint4*)(p0 + 16);
  return f.v;
}

// ---------------- Setup: ZOH discretization + f16 transposed weight copies ----
__global__ void ssm_setup_kernel(const float* __restrict__ raw_lambda,
                                 const float* __restrict__ B_c,
                                 const float* __restrict__ C,
                                 _Float16* __restrict__ BdT,   // [n][c]
                                 _Float16* __restrict__ CT,    // [o][n]
                                 float* __restrict__ a_arr,
                                 float* __restrict__ aL_arr) {
  const int row = blockIdx.x;
  const int tid = threadIdx.x;
  if (blockIdx.y == 0) {
    const int n = row;
    const float x  = raw_lambda[n];
    const float sp = (x > 20.f) ? x : log1pf(expf(x));   // softplus
    const float lam = -sp;
    const float a   = expf(lam);                          // A_d (DT = 1)
    const float factor = (fabsf(lam) > 1e-6f) ? (a - 1.f) / lam : 1.f;
    BdT[n * CIN + tid] = (_Float16)(B_c[tid * NST + n] * factor);
    if (tid == 0) { a_arr[n] = a; aL_arr[n] = expf((float)TCH * lam); }
  } else {
    const int o = row;
    CT[o * NST + tid] = (_Float16)(C[tid * COUT + o]);
  }
}

// ---------------- Fused: GEMM1 -> chunk scan -> chained carry -> GEMM2 -------
__global__ void __launch_bounds__(256)
ssm_fused_kernel(const float* __restrict__ u,
                 const _Float16* __restrict__ BdT,
                 const _Float16* __restrict__ CT,
                 const float* __restrict__ a_arr,
                 const float* __restrict__ aL_arr,
                 float* __restrict__ carry,     // [b][chunk][n] end-of-chunk state
                 int* __restrict__ flags,       // [b][chunk]
                 float* __restrict__ y) {
  __shared__ float     xbuf[TCH * XP];          // 67,584 B (f32 x / local-scan)
  __shared__ _Float16  sth [TCH * XP];          // 33,792 B (f16 corrected states)
  _Float16* uT  = (_Float16*)xbuf;              // overlay: 2 x 5 KB u stage buffers
  _Float16* ctb = (_Float16*)xbuf;              // overlay: 2 x 20 KB CT stage buffers

  const int chunk = blockIdx.x;
  const int b     = blockIdx.y;
  const int t0    = chunk * TCH;
  const int tid   = threadIdx.x;
  const int lane  = tid & 31;
  const int w     = tid >> 5;                   // wave 0..7
  const int r16   = lane & 15;
  const int hi    = lane >> 4;

  // ======== GEMM1: x[t][n] = sum_c u[b][c][t0+t] * Bd[c][n] ========
  const int tt    = w >> 1;                     // wave's t-tile (0..3)
  const int nbase = (w & 1) * 8;                // wave's 8 n-tiles
  v8f acc[8] = {};

  const int ci = tid & 31;                      // staging: c within slice
  const int tg = tid >> 5;                      // staging: t group (8 t's each)
  for (int ks = 0; ks < 8; ++ks) {
    const int c0 = ks * 32;
    // stage 64t x 32c slice of u as f16, t-major rows (double-buffered)
    _Float16* uTb = uT + (ks & 1) * UBUF;
    const float* up = u + (size_t)(b * CIN + c0 + ci) * T_LEN + t0 + tg * 8;
    float4 f0 = *(const float4*)(up);
    float4 f1 = *(const float4*)(up + 4);
    if (ks < 7) __builtin_prefetch(up + 32 * T_LEN, 0, 1);  // next K slice of u
    float vals[8] = {f0.x, f0.y, f0.z, f0.w, f1.x, f1.y, f1.z, f1.w};
#pragma unroll
    for (int j = 0; j < 8; ++j)
      uTb[(tg * 8 + j) * UP + ci] = (_Float16)vals[j];
    __syncthreads();

    const v16h afrag = ld_fragA(uTb + (tt * 16 + r16) * UP + hi * 8);
#pragma unroll
    for (int nt = 0; nt < 8; ++nt) {
      const int ncol = (nbase + nt) * 16 + r16;
      const v16h bfrag = ld_fragB(BdT + (size_t)ncol * CIN + c0 + hi * 8);
      acc[nt] = __builtin_amdgcn_wmma_f32_16x16x32_f16(
          false, afrag, false, bfrag, (short)0, acc[nt], false, false);
    }
  }
  __syncthreads();                              // uT dead; xbuf region reused
#pragma unroll
  for (int nt = 0; nt < 8; ++nt)
#pragma unroll
    for (int r = 0; r < 8; ++r)
      xbuf[(tt * 16 + r + 8 * hi) * XP + (nbase + nt) * 16 + r16] = acc[nt][r];
  __syncthreads();

  // ======== local scan along t (thread = n) ========
  const int n = tid;
  const float a = a_arr[n];
  float s = 0.f;
#pragma unroll 4
  for (int t = 0; t < TCH; ++t) {
    s = fmaf(a, s, xbuf[t * XP + n]);
    xbuf[t * XP + n] = s;                       // local prefix (zero init)
  }

  // ======== chained cross-chunk carry ========
  const int idx = b * NCHUNK + chunk;
  float cin = 0.f;
  if (chunk > 0) {
    if (tid == 0) {
      while (__hip_atomic_load(&flags[idx - 1], __ATOMIC_ACQUIRE,
                               __HIP_MEMORY_SCOPE_AGENT) == 0)
        __builtin_amdgcn_s_sleep(2);
    }
    __syncthreads();
    cin = carry[(size_t)(idx - 1) * NST + n];
  }
  carry[(size_t)idx * NST + n] = fmaf(aL_arr[n], cin, s);  // state at chunk end
  __threadfence();
  __syncthreads();
  if (tid == 0)
    __hip_atomic_store(&flags[idx], 1, __ATOMIC_RELEASE, __HIP_MEMORY_SCOPE_AGENT);

  // ======== correct with carry-in, convert to f16 ========
  float p = cin;                                // p -> a^(t+1) * cin
#pragma unroll 4
  for (int t = 0; t < TCH; ++t) {
    p *= a;
    sth[t * XP + n] = (_Float16)(xbuf[t * XP + n] + p);
  }
  __syncthreads();                              // sth ready; xbuf dead -> ctb

  // ======== GEMM2: y[b][o][t] = sum_n states[t][n] * C[n][o] (M=o,K=n,N=t) ====
  // Stage CT K-slices (256 o x 32 n, f16 = 16 KB) into LDS, double-buffered,
  // via gfx1250 async-to-LDS (ASYNCcnt) when available.
  auto stage_ct = [&](int ks, int buf) {
    const _Float16* src = CT + (size_t)tid * NST + ks * 32;   // row o = tid
    _Float16* dst = ctb + buf * CBUF + tid * CP;
#if HAVE_ASYNC_LDS
#pragma unroll
    for (int j = 0; j < 4; ++j)
      __builtin_amdgcn_global_load_async_to_lds_b128(
          (AS1 v4i_vec*)(src + j * 8), (AS3 v4i_vec*)(dst + j * 8), 0, 0);
#else
#pragma unroll
    for (int j = 0; j < 4; ++j)
      *(uint4*)(dst + j * 8) = *(const uint4*)(src + j * 8);
#endif
  };

  v8f acc2[2][4] = {};
  stage_ct(0, 0);
  for (int ks = 0; ks < 8; ++ks) {
    const int kb = ks * 32;
#if HAVE_ASYNC_LDS
    __builtin_amdgcn_s_wait_asynccnt(0);        // slice ks landed in LDS
#endif
    __syncthreads();
    if (ks < 7) stage_ct(ks + 1, (ks + 1) & 1); // overlap copy with compute
    const _Float16* cslice = ctb + (ks & 1) * CBUF;
    v16h afr0 = ld_fragA(cslice + ((2 * w + 0) * 16 + r16) * CP + hi * 8);
    v16h afr1 = ld_fragA(cslice + ((2 * w + 1) * 16 + r16) * CP + hi * 8);
#pragma unroll
    for (int ttile = 0; ttile < 4; ++ttile) {
      const v16h bfr = ld_fragB(sth + (ttile * 16 + r16) * XP + kb + hi * 8);
      acc2[0][ttile] = __builtin_amdgcn_wmma_f32_16x16x32_f16(
          false, afr0, false, bfr, (short)0, acc2[0][ttile], false, false);
      acc2[1][ttile] = __builtin_amdgcn_wmma_f32_16x16x32_f16(
          false, afr1, false, bfr, (short)0, acc2[1][ttile], false, false);
    }
    __syncthreads();
  }
#pragma unroll
  for (int ot = 0; ot < 2; ++ot)
#pragma unroll
    for (int ttile = 0; ttile < 4; ++ttile)
#pragma unroll
      for (int r = 0; r < 8; ++r) {
        const int o   = (2 * w + ot) * 16 + r + 8 * hi;
        const int tg2 = t0 + ttile * 16 + r16;
        y[(size_t)(b * COUT + o) * T_LEN + tg2] = acc2[ot][ttile][r];
      }
}

extern "C" void kernel_launch(void* const* d_in, const int* in_sizes, int n_in,
                              void* d_out, int out_size, void* d_ws, size_t ws_size,
                              hipStream_t stream) {
  (void)in_sizes; (void)n_in; (void)out_size; (void)ws_size;
  const float* u          = (const float*)d_in[0];
  const float* raw_lambda = (const float*)d_in[1];
  const float* B_c        = (const float*)d_in[2];
  const float* C          = (const float*)d_in[3];
  float* y = (float*)d_out;

  char* ws = (char*)d_ws;
  _Float16* BdT  = (_Float16*)(ws);                       // 131072 B
  _Float16* CT   = (_Float16*)(ws + 131072);              // 131072 B
  float*    a_a  = (float*)(ws + 262144);                 //   1024 B
  float*    aL_a = (float*)(ws + 263168);                 //   1024 B
  float*    carry= (float*)(ws + 264192);                 // 16*128*256*4 = 2 MiB
  int*      flags= (int*)  (ws + 264192 + 2097152);       //   8192 B

  (void)hipMemsetAsync(flags, 0, B_SZ * NCHUNK * sizeof(int), stream);
  ssm_setup_kernel<<<dim3(256, 2), 256, 0, stream>>>(raw_lambda, B_c, C,
                                                     BdT, CT, a_a, aL_a);
  ssm_fused_kernel<<<dim3(NCHUNK, B_SZ), 256, 0, stream>>>(u, BdT, CT, a_a, aL_a,
                                                           carry, flags, y);
}
